// DGN_54107997995605
// MI455X (gfx1250) — matrix operand order repaired and checked by reading, not compiled, not run on hardware
//
#include <hip/hip_runtime.h>

#define DEV __device__ __forceinline__

typedef __attribute__((ext_vector_type(16))) __bf16 v16bf;
typedef __attribute__((ext_vector_type(8)))  float  v8f;
typedef __attribute__((ext_vector_type(4)))  unsigned int u32x4;
typedef __attribute__((ext_vector_type(4)))  float  f32x4;

union BV { v16bf v; unsigned short us[16]; u32x4 q[2]; };

constexpr int Bc  = 32;     // batch
constexpr int Nn  = 1000;   // nodes
constexpr int DIN = 64;
constexpr int Hc  = 128;
constexpr int Ac  = 10;
constexpr int NP  = 1024;   // padded N (power of two, covers 63 row tiles + slack)
constexpr int NT  = 63;     // 16-row tiles covering 1008 >= 1000 rows
constexpr int M   = Bc * Nn;        // 32000
constexpr long MPl = (long)Bc * NP; // padded rows

// Native fp32 -> bf16 (lowered to v_cvt_pk_bf16_f32 on gfx1250)
DEV unsigned short f2bfu(float f) {
  __bf16 h = (__bf16)f;
  return __builtin_bit_cast(unsigned short, h);
}

DEV v8f wmma_bf16(const BV& a, const BV& b, v8f c) {
  return __builtin_amdgcn_wmma_f32_16x16x32_bf16(false, a.v, false, b.v,
                                                 (short)0, c, false, false);
}

// A-fragment 16x32 bf16 (row-major source): lane pair (ln, half) holds row ln,
// K = {half*8 .. +7} in q[0], K = {16+half*8 .. +7} in q[1].
DEV void load_a_frag(const unsigned short* rowp, int half, BV& a) {
  const unsigned short* p = rowp + half * 8;
  a.q[0] = *(const u32x4*)(p);
  a.q[1] = *(const u32x4*)(p + 16);
}
// B-fragment 32x16 bf16: lane = column; K = {half*8 .. +7} in q[0], K = {16+half*8 .. +7} in q[1].
DEV void load_b_frag(const unsigned short* colp, int half, BV& b) {
  const unsigned short* p = colp + half * 8;
  b.q[0] = *(const u32x4*)(p);
  b.q[1] = *(const u32x4*)(p + 16);
}

// ---------------------------------------------------------------------------
// Generic WMMA GEMM: out[b, r, c] = act(A[b, r, :] @ Wt[c, :] + bias[c])
// A fp32 row-major, Wt bf16 [128, K] row-major (pre-transposed weights).
// One wave computes a 16-row x 128-col tile.  outF (fp32) / outB (bf16)
// optional; output uses padded batch stride o_bstride with leading dim o_ld.
// ---------------------------------------------------------------------------
__global__ void __launch_bounds__(128)
gemm_bias_act(const float* __restrict__ A, int a_bstride, int a_valid, int K,
              const unsigned short* __restrict__ Wt, const float* __restrict__ bias,
              float* __restrict__ outF, unsigned short* __restrict__ outB,
              int o_bstride, int o_ld, int o_coloff, int do_relu, int n_tiles)
{
  const int wid = threadIdx.x >> 5, lane = threadIdx.x & 31;
  const int tile = blockIdx.x * 4 + wid;
  if (tile >= n_tiles) return;
  const int b = blockIdx.y;
  const int half = lane >> 4, ln = lane & 15;
  int r = tile * 16 + ln;
  int ar = (r < a_valid) ? r : (a_valid - 1);   // clamp tail rows
  const float* Arow = A + ((size_t)b * a_bstride + ar) * K;

  v8f acc[8] = {};
  for (int k0 = 0; k0 < K; k0 += 32) {
    BV a;
    const int hs = half * 8;
    f32x4 f0 = *(const f32x4*)(Arow + k0 + hs);
    f32x4 f1 = *(const f32x4*)(Arow + k0 + hs + 4);
    f32x4 f2 = *(const f32x4*)(Arow + k0 + 16 + hs);
    f32x4 f3 = *(const f32x4*)(Arow + k0 + 16 + hs + 4);
#pragma unroll
    for (int j = 0; j < 4; ++j) {
      a.v[j]      = (__bf16)f0[j];
      a.v[4 + j]  = (__bf16)f1[j];
      a.v[8 + j]  = (__bf16)f2[j];
      a.v[12 + j] = (__bf16)f3[j];
    }
#pragma unroll
    for (int ct = 0; ct < 8; ++ct) {
      BV bt;
      load_b_frag(Wt + (size_t)(ct * 16 + ln) * K + k0, half, bt);
      acc[ct] = wmma_bf16(a, bt, acc[ct]);
    }
  }
#pragma unroll
  for (int ct = 0; ct < 8; ++ct) {
    const float bv = bias[ct * 16 + ln];
#pragma unroll
    for (int i = 0; i < 8; ++i) {
      float v = acc[ct][i] + bv;
      if (do_relu) v = fmaxf(v, 0.0f);
      const int row = tile * 16 + i + half * 8;
      const size_t o = ((size_t)b * o_bstride + row) * o_ld + o_coloff + ct * 16 + ln;
      if (outF) outF[o] = v;
      if (outB) outB[o] = f2bfu(v);
    }
  }
}

// ---------------------------------------------------------------------------
// Flash attention: out = softmax(q k^T * mask - 9e15*(1-mask)) @ v
// One wave owns 16 query rows; loops over keys 32 at a time with online
// softmax.  P is re-tiled D-layout -> A-layout through LDS.
// ---------------------------------------------------------------------------
__global__ void __launch_bounds__(128)
flash_attn(const unsigned short* __restrict__ Qb, const unsigned short* __restrict__ Kb,
           const unsigned short* __restrict__ Vt, const float* __restrict__ mask,
           float* __restrict__ Out)
{
  const int wid = threadIdx.x >> 5, lane = threadIdx.x & 31;
  const int tile = blockIdx.x * 4 + wid;
  __shared__ unsigned short Pl[4][16 * 32];
  if (tile >= NT) return;
  const int b = blockIdx.y;
  const int r0 = tile * 16;
  const int half = lane >> 4, ln = lane & 15;
  unsigned short* P = Pl[wid];

  BV qa[4];
  {
    const unsigned short* qrow = Qb + ((size_t)b * NP + r0 + ln) * Hc;
#pragma unroll
    for (int kk = 0; kk < 4; ++kk) load_a_frag(qrow + kk * 32, half, qa[kk]);
  }
  v8f o[8] = {};
  float mrow[8], lrow[8];
#pragma unroll
  for (int i = 0; i < 8; ++i) { mrow[i] = -3.0e38f; lrow[i] = 0.0f; }

  for (int c0 = 0; c0 < NP; c0 += 32) {
    if (c0 + 32 < NP)
      __builtin_prefetch(Kb + ((size_t)b * NP + c0 + 32 + lane) * Hc, 0, 1);
    v8f s0 = {}, s1 = {};
    {
      const unsigned short* k0p = Kb + ((size_t)b * NP + c0 + ln) * Hc;
      const unsigned short* k1p = k0p + 16 * Hc;
#pragma unroll
      for (int kk = 0; kk < 4; ++kk) {
        BV bt;
        load_b_frag(k0p + kk * 32, half, bt);
        s0 = wmma_bf16(qa[kk], bt, s0);
        load_b_frag(k1p + kk * 32, half, bt);
        s1 = wmma_bf16(qa[kk], bt, s1);
      }
    }
#pragma unroll
    for (int i = 0; i < 8; ++i) {
      const int row = r0 + i + half * 8;
      const int ca = c0 + ln, cb = ca + 16;
      const float* mrowp = mask + ((size_t)b * Nn + (row < Nn ? row : 0)) * Nn;
      const float m0 = (row < Nn && ca < Nn) ? mrowp[ca] : 0.0f;
      const float m1 = (row < Nn && cb < Nn) ? mrowp[cb] : 0.0f;
      float sv0 = s0[i] * m0 - 9.0e15f * (1.0f - m0);
      float sv1 = s1[i] * m1 - 9.0e15f * (1.0f - m1);
      float tm = fmaxf(sv0, sv1);
#pragma unroll
      for (int off = 1; off < 16; off <<= 1) tm = fmaxf(tm, __shfl_xor(tm, off, 32));
      const float mnew  = fmaxf(mrow[i], tm);
      const float scale = __expf(mrow[i] - mnew);
      const float p0 = __expf(sv0 - mnew);
      const float p1 = __expf(sv1 - mnew);
      float rs = p0 + p1;
#pragma unroll
      for (int off = 1; off < 16; off <<= 1) rs += __shfl_xor(rs, off, 32);
      lrow[i] = lrow[i] * scale + rs;
      mrow[i] = mnew;
#pragma unroll
      for (int t = 0; t < 8; ++t) o[t][i] *= scale;
      P[(i + half * 8) * 32 + ln]      = f2bfu(p0);
      P[(i + half * 8) * 32 + ln + 16] = f2bfu(p1);
    }
    asm volatile("s_wait_dscnt 0" ::: "memory");   // LDS RAW within wave
    BV pa;
    load_a_frag(P + ln * 32, half, pa);
    const unsigned short* vbase = Vt + ((size_t)b * Hc) * NP + c0;
#pragma unroll
    for (int ct = 0; ct < 8; ++ct) {
      BV bvf;
      load_b_frag(vbase + (size_t)(ct * 16 + ln) * NP, half, bvf);
      o[ct] = wmma_bf16(pa, bvf, o[ct]);
    }
  }
#pragma unroll
  for (int i = 0; i < 8; ++i) {
    const int row = r0 + i + half * 8;
    const float inv = 1.0f / lrow[i];
#pragma unroll
    for (int ct = 0; ct < 8; ++ct)
      Out[((size_t)b * NP + row) * Hc + ct * 16 + ln] = o[ct][i] * inv;
  }
}

// ---------------------------------------------------------------------------
// Small helper kernels
// ---------------------------------------------------------------------------
__global__ void wconvT_kernel(const float* __restrict__ W, int K, int Nout,
                              unsigned short* __restrict__ Wt) {
  int idx = blockIdx.x * blockDim.x + threadIdx.x;
  if (idx >= K * Nout) return;
  int n = idx / K, k = idx % K;
  Wt[idx] = f2bfu(W[(size_t)k * Nout + n]);   // Wt[n,k] = W[k,n]
}

__global__ void wconvC_kernel(const float* __restrict__ W, int count,
                              unsigned short* __restrict__ out) {
  int idx = blockIdx.x * blockDim.x + threadIdx.x;
  if (idx < count) out[idx] = f2bfu(W[idx]);
}

__global__ void vtrans_kernel(const unsigned short* __restrict__ vB,
                              unsigned short* __restrict__ vtB) {
  size_t idx = (size_t)blockIdx.x * blockDim.x + threadIdx.x;
  const size_t total = (size_t)Bc * Hc * NP;
  if (idx >= total) return;
  int n = (int)(idx % NP);
  int h = (int)((idx / NP) % Hc);
  int b = (int)(idx / ((size_t)NP * Hc));
  unsigned short v = 0;
  if (n < Nn) v = vB[((size_t)b * NP + n) * Hc + h];
  vtB[idx] = v;                                // Vt[b,h,n]
}

__global__ void gru_kernel(const float* __restrict__ gx, const float* __restrict__ gh,
                           const float* __restrict__ h_in, float* __restrict__ out) {
  size_t idx = (size_t)blockIdx.x * blockDim.x + threadIdx.x;
  if (idx >= (size_t)M * Hc) return;
  int c = (int)(idx % Hc);
  int m = (int)(idx / Hc);
  int b = m / Nn, n = m % Nn;
  size_t pr = ((size_t)b * NP + n) * 384;
  float xr = gx[pr + c], xz = gx[pr + 128 + c], xn = gx[pr + 256 + c];
  float hr = gh[pr + c], hz = gh[pr + 128 + c], hn = gh[pr + 256 + c];
  float hi = h_in[(size_t)m * Hc + c];
  float r  = 1.0f / (1.0f + __expf(-(xr + hr)));
  float z  = 1.0f / (1.0f + __expf(-(xz + hz)));
  float nn = tanhf(xn + r * hn);
  out[(size_t)M * Ac + (size_t)m * Hc + c] = (1.0f - z) * nn + z * hi;
}

__global__ void fc_kernel(const float* __restrict__ h, const float* __restrict__ fcW,
                          const float* __restrict__ fcb, float* __restrict__ qout) {
  int idx = blockIdx.x * blockDim.x + threadIdx.x;
  if (idx >= M * Ac) return;
  int a = idx % Ac, m = idx / Ac;
  const float* hr = h + (size_t)m * Hc;
  float s = fcb[a];
#pragma unroll 8
  for (int c = 0; c < Hc; ++c) s += hr[c] * fcW[c * Ac + a];
  qout[idx] = s;
}

// ---------------------------------------------------------------------------
extern "C" void kernel_launch(void* const* d_in, const int* in_sizes, int n_in,
                              void* d_out, int out_size, void* d_ws, size_t ws_size,
                              hipStream_t stream)
{
  (void)in_sizes; (void)n_in; (void)out_size; (void)ws_size;
  const float* x    = (const float*)d_in[0];
  const float* mask = (const float*)d_in[1];
  const float* hst  = (const float*)d_in[2];
  const float* encW = (const float*)d_in[3];  const float* encb = (const float*)d_in[4];
  const float* Wv1  = (const float*)d_in[5];  const float* bv1  = (const float*)d_in[6];
  const float* Wk1  = (const float*)d_in[7];  const float* bk1  = (const float*)d_in[8];
  const float* Wq1  = (const float*)d_in[9];  const float* bq1  = (const float*)d_in[10];
  const float* Wo1  = (const float*)d_in[11]; const float* bo1  = (const float*)d_in[12];
  const float* Wv2  = (const float*)d_in[13]; const float* bv2  = (const float*)d_in[14];
  const float* Wk2  = (const float*)d_in[15]; const float* bk2  = (const float*)d_in[16];
  const float* Wq2  = (const float*)d_in[17]; const float* bq2  = (const float*)d_in[18];
  const float* Wo2  = (const float*)d_in[19]; const float* bo2  = (const float*)d_in[20];
  const float* Wih  = (const float*)d_in[21]; const float* bih  = (const float*)d_in[22];
  const float* Whh  = (const float*)d_in[23]; const float* bhh  = (const float*)d_in[24];
  const float* fcW  = (const float*)d_in[25]; const float* fcb  = (const float*)d_in[26];
  float* out = (float*)d_out;

  char* ws = (char*)d_ws;
  size_t off = 0;
  auto take = [&](size_t bytes) {
    char* p = ws + off;
    off = (off + bytes + 255) & ~(size_t)255;
    return p;
  };
  float* h1F  = (float*)take((size_t)MPl * Hc * 4);      // also reused as h3
  float* attF = (float*)take((size_t)MPl * Hc * 4);
  float* h2F  = (float*)take((size_t)MPl * Hc * 4);
  unsigned short* qB  = (unsigned short*)take((size_t)MPl * Hc * 2);
  unsigned short* kB  = (unsigned short*)take((size_t)MPl * Hc * 2);
  unsigned short* vB  = (unsigned short*)take((size_t)MPl * Hc * 2);
  unsigned short* vtB = (unsigned short*)take((size_t)Bc * Hc * NP * 2);
  float* gx = (float*)take((size_t)MPl * 384 * 4);
  float* gh = (float*)take((size_t)MPl * 384 * 4);
  unsigned short* encWt = (unsigned short*)take((size_t)DIN * Hc * 2);
  unsigned short* Wq1t = (unsigned short*)take((size_t)Hc * Hc * 2);
  unsigned short* Wk1t = (unsigned short*)take((size_t)Hc * Hc * 2);
  unsigned short* Wv1t = (unsigned short*)take((size_t)Hc * Hc * 2);
  unsigned short* Wo1t = (unsigned short*)take((size_t)Hc * Hc * 2);
  unsigned short* Wq2t = (unsigned short*)take((size_t)Hc * Hc * 2);
  unsigned short* Wk2t = (unsigned short*)take((size_t)Hc * Hc * 2);
  unsigned short* Wv2t = (unsigned short*)take((size_t)Hc * Hc * 2);
  unsigned short* Wo2t = (unsigned short*)take((size_t)Hc * Hc * 2);
  unsigned short* WihB = (unsigned short*)take((size_t)384 * Hc * 2);
  unsigned short* WhhB = (unsigned short*)take((size_t)384 * Hc * 2);
  float* h3F = h1F;

  // --- weight prep (bf16, B-fragment-friendly layouts) ---
  wconvT_kernel<<<(DIN * Hc + 255) / 256, 256, 0, stream>>>(encW, DIN, Hc, encWt);
  wconvT_kernel<<<(Hc * Hc + 255) / 256, 256, 0, stream>>>(Wq1, Hc, Hc, Wq1t);
  wconvT_kernel<<<(Hc * Hc + 255) / 256, 256, 0, stream>>>(Wk1, Hc, Hc, Wk1t);
  wconvT_kernel<<<(Hc * Hc + 255) / 256, 256, 0, stream>>>(Wv1, Hc, Hc, Wv1t);
  wconvT_kernel<<<(Hc * Hc + 255) / 256, 256, 0, stream>>>(Wo1, Hc, Hc, Wo1t);
  wconvT_kernel<<<(Hc * Hc + 255) / 256, 256, 0, stream>>>(Wq2, Hc, Hc, Wq2t);
  wconvT_kernel<<<(Hc * Hc + 255) / 256, 256, 0, stream>>>(Wk2, Hc, Hc, Wk2t);
  wconvT_kernel<<<(Hc * Hc + 255) / 256, 256, 0, stream>>>(Wv2, Hc, Hc, Wv2t);
  wconvT_kernel<<<(Hc * Hc + 255) / 256, 256, 0, stream>>>(Wo2, Hc, Hc, Wo2t);
  wconvC_kernel<<<(384 * Hc + 255) / 256, 256, 0, stream>>>(Wih, 384 * Hc, WihB);
  wconvC_kernel<<<(384 * Hc + 255) / 256, 256, 0, stream>>>(Whh, 384 * Hc, WhhB);

  const dim3 gg((NT + 3) / 4, Bc);
  const size_t vtr_total = (size_t)Bc * Hc * NP;

  // --- encoder ---
  gemm_bias_act<<<gg, 128, 0, stream>>>(x, Nn, Nn, DIN, encWt, encb,
                                        h1F, nullptr, NP, Hc, 0, 1, NT);
  // --- attention layer 1 ---
  gemm_bias_act<<<gg, 128, 0, stream>>>(h1F, NP, NT * 16, Hc, Wq1t, bq1,
                                        nullptr, qB, NP, Hc, 0, 1, NT);
  gemm_bias_act<<<gg, 128, 0, stream>>>(h1F, NP, NT * 16, Hc, Wk1t, bk1,
                                        nullptr, kB, NP, Hc, 0, 1, NT);
  gemm_bias_act<<<gg, 128, 0, stream>>>(h1F, NP, NT * 16, Hc, Wv1t, bv1,
                                        nullptr, vB, NP, Hc, 0, 1, NT);
  vtrans_kernel<<<(unsigned)((vtr_total + 255) / 256), 256, 0, stream>>>(vB, vtB);
  flash_attn<<<gg, 128, 0, stream>>>(qB, kB, vtB, mask, attF);
  gemm_bias_act<<<gg, 128, 0, stream>>>(attF, NP, NT * 16, Hc, Wo1t, bo1,
                                        h2F, nullptr, NP, Hc, 0, 1, NT);
  // --- attention layer 2 ---
  gemm_bias_act<<<gg, 128, 0, stream>>>(h2F, NP, NT * 16, Hc, Wq2t, bq2,
                                        nullptr, qB, NP, Hc, 0, 1, NT);
  gemm_bias_act<<<gg, 128, 0, stream>>>(h2F, NP, NT * 16, Hc, Wk2t, bk2,
                                        nullptr, kB, NP, Hc, 0, 1, NT);
  gemm_bias_act<<<gg, 128, 0, stream>>>(h2F, NP, NT * 16, Hc, Wv2t, bv2,
                                        nullptr, vB, NP, Hc, 0, 1, NT);
  vtrans_kernel<<<(unsigned)((vtr_total + 255) / 256), 256, 0, stream>>>(vB, vtB);
  flash_attn<<<gg, 128, 0, stream>>>(qB, kB, vtB, mask, attF);
  gemm_bias_act<<<gg, 128, 0, stream>>>(attF, NP, NT * 16, Hc, Wo2t, bo2,
                                        h3F, nullptr, NP, Hc, 0, 1, NT);
  // --- GRU gate GEMMs (r, z, n), columns of [*,384] buffers ---
  for (int g = 0; g < 3; ++g) {
    gemm_bias_act<<<gg, 128, 0, stream>>>(h3F, NP, NT * 16, Hc,
                                          WihB + (size_t)g * 128 * Hc, bih + g * 128,
                                          gx, nullptr, NP, 384, g * 128, 0, NT);
    gemm_bias_act<<<gg, 128, 0, stream>>>(hst, Nn, Nn, Hc,
                                          WhhB + (size_t)g * 128 * Hc, bhh + g * 128,
                                          gh, nullptr, NP, 384, g * 128, 0, NT);
  }
  // --- GRU elementwise + output head ---
  gru_kernel<<<(unsigned)(((size_t)M * Hc + 255) / 256), 256, 0, stream>>>(gx, gh, hst, out);
  fc_kernel<<<(M * Ac + 255) / 256, 256, 0, stream>>>(out + (size_t)M * Ac, fcW, fcb, out);
}
